// DeepFactorMinerV5_13563506721347
// MI455X (gfx1250) — compile-verified, matrix operand order checked by reading, not run on hardware
//
#include <hip/hip_runtime.h>
#include <cmath>

typedef __attribute__((ext_vector_type(16))) _Float16 v16h;
typedef __attribute__((ext_vector_type(8)))  float    v8f;

#define B_SZ 16
#define T_SZ 1024
#define M_SZ (B_SZ * T_SZ)   // 16384 rows (b*T + t)
#define DM   256             // D_MODEL
#define DI   512             // D_INNER
#define DS_N 16              // D_STATE
#define DTR  16              // DT_RANK
#define NPROJ 512
#define NOUT  128
#define EPS_ 1e-5f

// GEMM tiling
#define MT 128
#define NT 64
#define KT 32

__device__ __forceinline__ float sigmoidf_(float x) { return 1.0f / (1.0f + __expf(-x)); }
__device__ __forceinline__ float siluf_(float x)    { return x * sigmoidf_(x); }
__device__ __forceinline__ float geluf_(float x)    { return 0.5f * x * (1.0f + erff(x * 0.70710678118654752f)); }
__device__ __forceinline__ float softplusf_(float x){ return (x > 20.0f) ? x : log1pf(__expf(x)); }

__device__ __forceinline__ void pack8(_Float16* dst, float4 f0, float4 f1) {
  dst[0] = (_Float16)f0.x; dst[1] = (_Float16)f0.y;
  dst[2] = (_Float16)f0.z; dst[3] = (_Float16)f0.w;
  dst[4] = (_Float16)f1.x; dst[5] = (_Float16)f1.y;
  dst[6] = (_Float16)f1.z; dst[7] = (_Float16)f1.w;
}

// ---------------------------------------------------------------------------
// WMMA GEMM: C[M_SZ, N] = bias[n] + A_eff[m, :K] . W_eff[n, :K]
//   MODE 0: A_eff[m,k] = A[m*lda + k], W_eff[n,k] = W[n*K + k]  (row-major NxK)
//   MODE 1: conv k=3 pad=1 over (B,T,CIN) with K permuted as k = dk*CIN + cin
//           (same permutation on both operands leaves the dot product intact):
//           A_eff[m,k] = X[b, t+dk-1, cin] (0 outside [0,T))
//           W_eff[n,k] = W[n, cin, dk] = W[n*K + cin*3 + dk]
// MODE/CIN are compile-time so the conv k->(dk,cin) split folds to shifts /
// one magic-multiply instead of a runtime division chain in the k-loop.
//
// Block: 256 threads / 8 waves. Tile 128x64, K-step 32. A/B staged in LDS in
// WMMA fragment order (per ISA 7.12.2: lane (r,hi) owns halves e=0..15 with
// k = {hi*8..+7, 16+hi*8..+7}), so a fragment load is one contiguous 32B LDS
// read. Each wave owns a 32x32 region = 4 accumulators = 4 v_wmma per k-step.
// ---------------------------------------------------------------------------
template <int MODE, int CIN>
__global__ __launch_bounds__(256)
void k_wmma_gemm(const float* __restrict__ A, const float* __restrict__ W,
                 const float* __restrict__ bias, float* __restrict__ C,
                 int N, int K, int lda)
{
  __shared__ __align__(32) _Float16 lA[MT * KT];  // [(row*2+hi)*16 + e]
  __shared__ __align__(32) _Float16 lB[NT * KT];

  const int tid  = threadIdx.x;
  const int lane = tid & 31;
  const int wave = tid >> 5;
  const int hi   = lane >> 4;
  const int r    = lane & 15;
  const int wm   = wave >> 1;      // 0..3 : M sub-block (32 rows)
  const int wn   = wave & 1;       // 0..1 : N sub-block (32 cols)

  const int mBlock = blockIdx.y * MT;
  const int nBlock = blockIdx.x * NT;

  // staging assignment: one (row, half) pair per thread
  const int  sRowA = tid >> 1;            // 0..127
  const int  sHi   = tid & 1;
  const int  sRowB = (tid & 127) >> 1;    // 0..63 (threads 0..127 stage B)
  const bool doB   = tid < 128;

  const int mRowA = mBlock + sRowA;
  const int bIdxA = mRowA >> 10;          // T_SZ == 1024
  const int tIdxA = mRowA & (T_SZ - 1);
  const int nRowB = nBlock + sRowB;

  v8f acc[2][2] = {};

  const int kSteps = (K + KT - 1) / KT;
  for (int ks = 0; ks < kSteps; ++ks) {
    const int kb = ks * KT;

    // ---- stage A fragment halves for (sRowA, sHi): two contiguous 8-runs
    {
      union { v16h v; _Float16 h[16]; } ua;
      #pragma unroll
      for (int run = 0; run < 2; ++run) {
        const int k0 = kb + run * 16 + sHi * 8;
        float4 f0 = {0.f, 0.f, 0.f, 0.f}, f1 = f0;
        if (k0 < K) {
          if (MODE == 0) {
            const float4* p = (const float4*)(A + (size_t)mRowA * lda + k0);
            f0 = p[0]; f1 = p[1];
          } else {
            const int dk  = k0 / CIN;               // permuted K: k = dk*CIN+cin
            const int cin = k0 - dk * CIN;          // runs never cross dk (CIN%8==0)
            const int tt  = tIdxA + dk - 1;
            if (tt >= 0 && tt < T_SZ) {
              const float4* p = (const float4*)(A + (size_t)(bIdxA * T_SZ + tt) * CIN + cin);
              f0 = p[0]; f1 = p[1];
            }
          }
        }
        pack8(&ua.h[run * 8], f0, f1);
      }
      *(v16h*)&lA[(sRowA * 2 + sHi) * 16] = ua.v;
      if (MODE == 0 && ks + 1 < kSteps)
        __builtin_prefetch(A + (size_t)mRowA * lda + kb + KT, 0, 1);
    }

    // ---- stage B fragment halves for (sRowB, sHi)
    if (doB) {
      union { v16h v; _Float16 h[16]; } ub;
      #pragma unroll
      for (int run = 0; run < 2; ++run) {
        const int k0 = kb + run * 16 + sHi * 8;
        if (nRowB < N && k0 < K) {
          if (MODE == 0) {
            const float4* p = (const float4*)(W + (size_t)nRowB * K + k0);
            pack8(&ub.h[run * 8], p[0], p[1]);
          } else {
            const int dk  = k0 / CIN;
            const int cin = k0 - dk * CIN;
            #pragma unroll
            for (int i = 0; i < 8; ++i)
              ub.h[run * 8 + i] = (_Float16)W[(size_t)nRowB * K + (cin + i) * 3 + dk];
          }
        } else {
          #pragma unroll
          for (int i = 0; i < 8; ++i) ub.h[run * 8 + i] = (_Float16)0.0f;
        }
      }
      *(v16h*)&lB[(sRowB * 2 + sHi) * 16] = ub.v;
      if (MODE == 0 && ks + 1 < kSteps)
        __builtin_prefetch(W + (size_t)nRowB * K + kb + KT, 0, 1);
    }
    __syncthreads();

    // ---- 4 fragments -> 4 WMMAs (A reused 2x, B reused 2x within the wave)
    const v16h a0 = *(const v16h*)&lA[((wm * 32 +  0 + r) * 2 + hi) * 16];
    const v16h a1 = *(const v16h*)&lA[((wm * 32 + 16 + r) * 2 + hi) * 16];
    const v16h b0 = *(const v16h*)&lB[((wn * 32 +  0 + r) * 2 + hi) * 16];
    const v16h b1 = *(const v16h*)&lB[((wn * 32 + 16 + r) * 2 + hi) * 16];

    acc[0][0] = __builtin_amdgcn_wmma_f32_16x16x32_f16(false, a0, false, b0, (short)0, acc[0][0], false, false);
    acc[0][1] = __builtin_amdgcn_wmma_f32_16x16x32_f16(false, a0, false, b1, (short)0, acc[0][1], false, false);
    acc[1][0] = __builtin_amdgcn_wmma_f32_16x16x32_f16(false, a1, false, b0, (short)0, acc[1][0], false, false);
    acc[1][1] = __builtin_amdgcn_wmma_f32_16x16x32_f16(false, a1, false, b1, (short)0, acc[1][1], false, false);
    __syncthreads();
  }

  // ---- store: lane column = r, rows rr + 8*hi (ISA C/D layout)
  #pragma unroll
  for (int mi = 0; mi < 2; ++mi) {
    #pragma unroll
    for (int ni = 0; ni < 2; ++ni) {
      const int nCol = nBlock + wn * 32 + ni * 16 + r;
      if (nCol < N) {
        const float bn = bias[nCol];
        union { v8f v; float f[8]; } res; res.v = acc[mi][ni];
        #pragma unroll
        for (int rr = 0; rr < 8; ++rr) {
          const int m = mBlock + wm * 32 + mi * 16 + rr + 8 * hi;
          C[(size_t)m * N + nCol] = res.f[rr] + bn;
        }
      }
    }
  }
}

// ---------------------------------------------------------------------------
// BatchNorm statistics over (B,T) per channel, coalesced: 32 channels/block,
// lanes on consecutive channels. stats[c]=mean, stats[DM+c]=invstd.
// ---------------------------------------------------------------------------
__global__ __launch_bounds__(256)
void k_bn_stats(const float* __restrict__ X, float* __restrict__ stats)
{
  const int c0   = blockIdx.x * 32;
  const int lane = threadIdx.x & 31;
  const int grp  = threadIdx.x >> 5;       // 8 row groups
  const int c    = c0 + lane;
  __shared__ float s1[256], s2[256];

  float sum = 0.0f, sq = 0.0f;
  for (int m = grp; m < M_SZ; m += 8) {
    const float v = X[(size_t)m * DM + c];
    sum += v; sq += v * v;
  }
  s1[threadIdx.x] = sum; s2[threadIdx.x] = sq;
  __syncthreads();
  for (int g = 4; g > 0; g >>= 1) {
    if (threadIdx.x < g * 32) {
      s1[threadIdx.x] += s1[threadIdx.x + g * 32];
      s2[threadIdx.x] += s2[threadIdx.x + g * 32];
    }
    __syncthreads();
  }
  if (threadIdx.x < 32) {
    const float mean = s1[threadIdx.x] * (1.0f / (float)M_SZ);
    const float var  = s2[threadIdx.x] * (1.0f / (float)M_SZ) - mean * mean;
    stats[c]      = mean;
    stats[DM + c] = rsqrtf(var + EPS_);
  }
}

__global__ void k_bn_silu(const float* __restrict__ X, const float* __restrict__ stats,
                          const float* __restrict__ g, const float* __restrict__ be,
                          float* __restrict__ Y)
{
  const int idx = blockIdx.x * blockDim.x + threadIdx.x;
  if (idx >= M_SZ * DM) return;
  const int c = idx & (DM - 1);
  const float v = (X[idx] - stats[c]) * stats[DM + c] * g[c] + be[c];
  Y[idx] = siluf_(v);
}

// ---------------------------------------------------------------------------
// Selective-SSM scan. One block per batch, one lane per inner channel d.
// State h[16], A[d,:], Wdt[d,:] live in VGPRs; per-step (dtr|B|C) row (48
// floats) broadcast via LDS. Wdt projection + softplus fused per step.
// Emits gated output only at t = T-1 (the only timestep consumed downstream).
// ---------------------------------------------------------------------------
__global__ __launch_bounds__(512)
void k_scan(const float* __restrict__ xz, const float* __restrict__ dbc,
            const float* __restrict__ A_log, const float* __restrict__ Dp,
            const float* __restrict__ Wdt, const float* __restrict__ bdt,
            float* __restrict__ ylast)
{
  const int d = threadIdx.x;
  const int b = blockIdx.x;
  float Ar[DS_N], Wd[DTR], h[DS_N];
  #pragma unroll
  for (int n = 0; n < DS_N; ++n) {
    Ar[n] = -__expf(A_log[d * DS_N + n]);
    Wd[n] = Wdt[d * DTR + n];
    h[n]  = 0.0f;
  }
  const float bd  = bdt[d];
  const float Dpd = Dp[d];
  __shared__ float row[48];

  for (int t = 0; t < T_SZ; ++t) {
    const size_t base = (size_t)(b * T_SZ + t);
    if (threadIdx.x < 48) row[threadIdx.x] = dbc[base * 48 + threadIdx.x];
    __syncthreads();

    float acc = bd;
    #pragma unroll
    for (int r = 0; r < DTR; ++r) acc += Wd[r] * row[r];
    const float dt  = softplusf_(acc);
    const float xt  = xz[base * 1024 + d];
    const float dtx = dt * xt;
    float y = 0.0f;
    #pragma unroll
    for (int n = 0; n < DS_N; ++n) {
      const float dA = __expf(dt * Ar[n]);
      h[n] = h[n] * dA + dtx * row[16 + n];
      y += h[n] * row[32 + n];
    }
    if (t == T_SZ - 1) {
      const float z = xz[base * 1024 + DI + d];
      ylast[b * DI + d] = (y + xt * Dpd) * siluf_(z);
    }
    __syncthreads();
  }
}

// ---------------------------------------------------------------------------
// Tail: Wout (both streams) -> FiLM -> mixer(GELU) -> proj/LayerNorm/GELU ->
// output. Operates only on the final timestep: 16 rows, all staged in LDS.
// ---------------------------------------------------------------------------
__global__ __launch_bounds__(256)
void k_tail(const float* __restrict__ yT, const float* __restrict__ yA,
            const float* __restrict__ WoT, const float* __restrict__ boT,
            const float* __restrict__ WoA, const float* __restrict__ boA,
            const float* __restrict__ mask,
            const float* __restrict__ Ws,  const float* __restrict__ bs,
            const float* __restrict__ Wsh, const float* __restrict__ bsh,
            const float* __restrict__ mW1, const float* __restrict__ mb1,
            const float* __restrict__ mW2, const float* __restrict__ mb2,
            const float* __restrict__ pW1, const float* __restrict__ pb1,
            const float* __restrict__ pg,  const float* __restrict__ pbe,
            const float* __restrict__ pW2, const float* __restrict__ pb2,
            float* __restrict__ out)
{
  __shared__ float sA[B_SZ * NPROJ];   // 16x512 staging
  __shared__ float sB[B_SZ * DM];      // 16x256 staging
  __shared__ float sMV[2 * B_SZ];
  const int tid = threadIdx.x;

  // h_tgt -> sA[0..4095], h_aux -> sA[4096..8191]
  for (int i = tid; i < B_SZ * DM; i += 256) {
    const int b = i >> 8, c = i & 255;
    float a0 = boT[c], a1 = boA[c];
    for (int k = 0; k < DI; ++k) {
      a0 += yT[b * DI + k] * WoT[c * DI + k];
      a1 += yA[b * DI + k] * WoA[c * DI + k];
    }
    sA[i] = a0; sA[B_SZ * DM + i] = a1;
  }
  __syncthreads();

  // FiLM -> sB : tgt*(1+gamma)+beta, gamma/beta linear in (aux*mask)
  for (int i = tid; i < B_SZ * DM; i += 256) {
    const int b = i >> 8, c = i & 255;
    const float mk = mask[b * T_SZ + (T_SZ - 1)];
    const float* ha = &sA[B_SZ * DM + b * DM];
    float d1 = 0.0f, d2 = 0.0f;
    for (int j = 0; j < DM; ++j) {
      const float v = ha[j];
      d1 += Ws[c * DM + j]  * v;
      d2 += Wsh[c * DM + j] * v;
    }
    const float gm  = bs[c]  + mk * d1;
    const float bt2 = bsh[c] + mk * d2;
    sB[i] = sA[b * DM + c] * (1.0f + gm) + bt2;
  }
  __syncthreads();

  // mixer layer 1 (256->512) + exact GELU -> sA
  for (int i = tid; i < B_SZ * NPROJ; i += 256) {
    const int b = i >> 9, n = i & 511;
    float a = mb1[n];
    for (int j = 0; j < DM; ++j) a += mW1[n * DM + j] * sB[b * DM + j];
    sA[i] = geluf_(a);
  }
  __syncthreads();

  // mixer layer 2 (512->256) -> sB
  for (int i = tid; i < B_SZ * DM; i += 256) {
    const int b = i >> 8, c = i & 255;
    float a = mb2[c];
    for (int n = 0; n < NPROJ; ++n) a += mW2[c * NPROJ + n] * sA[b * NPROJ + n];
    sB[i] = a;
  }
  __syncthreads();

  // proj layer 1 (256->512) -> sA
  for (int i = tid; i < B_SZ * NPROJ; i += 256) {
    const int b = i >> 9, n = i & 511;
    float a = pb1[n];
    for (int j = 0; j < DM; ++j) a += pW1[n * DM + j] * sB[b * DM + j];
    sA[i] = a;
  }
  __syncthreads();

  // per-row LayerNorm stats
  if (tid < B_SZ) {
    float s = 0.0f, q = 0.0f;
    for (int n = 0; n < NPROJ; ++n) { const float v = sA[tid * NPROJ + n]; s += v; q += v * v; }
    const float mean = s * (1.0f / (float)NPROJ);
    const float var  = q * (1.0f / (float)NPROJ) - mean * mean;
    sMV[tid]        = mean;
    sMV[B_SZ + tid] = rsqrtf(var + EPS_);
  }
  __syncthreads();

  // normalize + affine + GELU in place
  for (int i = tid; i < B_SZ * NPROJ; i += 256) {
    const int b = i >> 9, n = i & 511;
    const float v = (sA[i] - sMV[b]) * sMV[B_SZ + b] * pg[n] + pbe[n];
    sA[i] = geluf_(v);
  }
  __syncthreads();

  // proj layer 2 (512->128) -> out
  for (int i = tid; i < B_SZ * NOUT; i += 256) {
    const int b = i >> 7, o = i & 127;
    float a = pb2[o];
    for (int n = 0; n < NPROJ; ++n) a += pW2[o * NPROJ + n] * sA[b * NPROJ + n];
    out[i] = a;
  }
}

// ---------------------------------------------------------------------------
extern "C" void kernel_launch(void* const* d_in, const int* in_sizes, int n_in,
                              void* d_out, int out_size, void* d_ws, size_t ws_size,
                              hipStream_t stream)
{
  (void)in_sizes; (void)n_in; (void)out_size; (void)ws_size;
  const float* x_tgt = (const float*)d_in[0];
  const float* x_aux = (const float*)d_in[1];
  const float* amask = (const float*)d_in[2];
  auto P = [&](int i) { return (const float*)d_in[i]; };

  char* ws = (char*)d_ws;
  const size_t MB = 1ull << 20;
  float* convbuf = (float*)(ws);               // 16 MB  (conv pre-BN; reused as enc_out in-place)
  float* act1    = (float*)(ws + 16 * MB);     // 16 MB  (post conv1 BN+SiLU)
  float* xz      = (float*)(ws + 32 * MB);     // 64 MB  (Win output, x|z)
  float* dbc     = (float*)(ws + 96 * MB);     // ~3 MB  (dtr|B|C)
  float* stats   = (float*)(ws + 96 * MB + (size_t)M_SZ * 48 * 4);
  float* yT      = (float*)((char*)stats + 4096);
  float* yA      = yT + B_SZ * DI;

  auto run_stream = [&](const float* xin, int pe, int ps, float* ylast) {
    const dim3 blk(256);
    const int elemBlocks = (M_SZ * DM + 255) / 256;
    // conv1 (im2col GEMM: N=256, K=40*3) -> convbuf
    k_wmma_gemm<1, 40><<<dim3((DM + NT - 1) / NT, M_SZ / MT), blk, 0, stream>>>(
        xin, P(pe + 0), P(pe + 1), convbuf, DM, 40 * 3, 0);
    k_bn_stats<<<DM / 32, 256, 0, stream>>>(convbuf, stats);
    k_bn_silu<<<elemBlocks, 256, 0, stream>>>(convbuf, stats, P(pe + 2), P(pe + 3), act1);
    // conv2 (N=256, K=256*3) -> convbuf
    k_wmma_gemm<1, 256><<<dim3((DM + NT - 1) / NT, M_SZ / MT), blk, 0, stream>>>(
        act1, P(pe + 4), P(pe + 5), convbuf, DM, DM * 3, 0);
    k_bn_stats<<<DM / 32, 256, 0, stream>>>(convbuf, stats);
    k_bn_silu<<<elemBlocks, 256, 0, stream>>>(convbuf, stats, P(pe + 6), P(pe + 7), convbuf);
    // Win: (B*T,256) -> (B*T,1024) = x|z
    k_wmma_gemm<0, 0><<<dim3(1024 / NT, M_SZ / MT), blk, 0, stream>>>(
        convbuf, P(ps + 0), P(ps + 1), xz, 1024, DM, DM);
    // Wx: x part (lda=1024, K=512) -> dbc (B*T,48)
    k_wmma_gemm<0, 0><<<dim3(1, M_SZ / MT), blk, 0, stream>>>(
        xz, P(ps + 2), P(ps + 3), dbc, 48, DI, 1024);
    // recurrent scan (fused Wdt+softplus), emit gated y at t=T-1
    k_scan<<<B_SZ, DI, 0, stream>>>(xz, dbc, P(ps + 4), P(ps + 5), P(ps + 6), P(ps + 7), ylast);
  };

  run_stream(x_tgt, 3, 19, yT);   // enc_tgt params @3..10, ssm_tgt @19..28
  run_stream(x_aux, 11, 29, yA);  // enc_aux params @11..18, ssm_aux @29..38

  k_tail<<<1, 256, 0, stream>>>(
      yT, yA, P(27), P(28), P(37), P(38), amask,
      P(39), P(40), P(41), P(42),          // film
      P(43), P(44), P(45), P(46),          // mixer
      P(47), P(48), P(49), P(50), P(51), P(52),  // proj
      (float*)d_out);
}
